// TransformerDecoder_37812892074571
// MI455X (gfx1250) — compile-verified, matrix operand order checked by reading, not run on hardware
//
#include <hip/hip_runtime.h>
#include <hip/hip_bf16.h>
#include <math.h>

// ---------------- model constants ----------------
#define cL 2
#define cE 8
#define cD 1024
#define cFF 2048
#define cH 16
#define cB 2
#define cS 1024
#define TOKENS (cB * cS)           // 2048
#define BSD (TOKENS * cD)          // 2,097,152
#define DH (cD / cH)               // 64

// ---------------- gfx1250 async global->LDS support (probe via __has_builtin) ----------------
#if __has_builtin(__builtin_amdgcn_global_load_async_to_lds_b128) && \
    __has_builtin(__builtin_amdgcn_s_wait_asynccnt)
#define USE_ASYNC_LDS 1
#else
#define USE_ASYNC_LDS 0
#endif

#define AS_GLOBAL __attribute__((address_space(1)))
#define AS_LDS    __attribute__((address_space(3)))

// ---------------- types / helpers ----------------
typedef __attribute__((ext_vector_type(16))) __bf16 bf16x16;
typedef __attribute__((ext_vector_type(8)))  float  f32x8;
typedef int v4i __attribute__((vector_size(16)));   // matches builtin param type

struct __attribute__((aligned(16))) U128 { unsigned int v[4]; };
struct B32x8 { U128 lo, hi; };

__device__ __forceinline__ unsigned short f2bf(float f) {
    union { float f; unsigned u; } v; v.f = f;
    unsigned r = v.u + 0x7fffu + ((v.u >> 16) & 1u);   // round-to-nearest-even
    return (unsigned short)(r >> 16);
}
__device__ __forceinline__ float bf2f(unsigned short u) {
    union { unsigned u; float f; } v; v.u = ((unsigned)u) << 16; return v.f;
}
// Load one WMMA 16-lane fragment row (16 bf16) from LDS: two contiguous 16B chunks.
__device__ __forceinline__ bf16x16 frag_ld(const unsigned short* p) {
    B32x8 u;
    u.lo = *(const U128*)(p);
    u.hi = *(const U128*)(p + 16);
    return __builtin_bit_cast(bf16x16, u);
}
// 32-byte global -> LDS copy (async on gfx1250 toolchains that expose the builtin).
__device__ __forceinline__ void copy32_g2l(const unsigned short* g, unsigned short* s) {
#if USE_ASYNC_LDS
    __builtin_amdgcn_global_load_async_to_lds_b128(
        (AS_GLOBAL v4i*)g, (AS_LDS v4i*)s, 0, 0);
    __builtin_amdgcn_global_load_async_to_lds_b128(
        (AS_GLOBAL v4i*)(g + 8), (AS_LDS v4i*)(s + 8), 0, 0);
#else
    U128 lo = *(const U128*)g, hi = *(const U128*)(g + 8);
    ((U128*)s)[0] = lo; ((U128*)s)[1] = hi;
#endif
}
__device__ __forceinline__ void wait_async_copies() {
#if USE_ASYNC_LDS
    __builtin_amdgcn_s_wait_asynccnt(0);
#endif
}

// ---------------- f32 -> bf16 conversion ----------------
__global__ __launch_bounds__(256) void cvt_f32_to_bf16_k(const float* __restrict__ src,
                                                         unsigned short* __restrict__ dst,
                                                         size_t n) {
    size_t i = (size_t)blockIdx.x * blockDim.x + threadIdx.x;
    size_t stride = (size_t)gridDim.x * blockDim.x;
    for (; i < n; i += stride) dst[i] = f2bf(src[i]);
}

// ---------------- bf16 WMMA GEMM, double-buffered LDS pipeline ----------------
// C[z] = epilogue( alpha * A[z] @ op(B[z]) + bias )  (op = transpose if TRANSB)
// Batch offsets: off = (z / zdiv) * outer + (z % zdiv) * inner  (element units)
// Block: 256 threads = 8 wave32; tile 128 x BN; K step 32; 2-deep LDS ping-pong.
template <int BN, bool TRANSB, bool GELU, bool STOREBF, bool ACCUM>
__global__ __launch_bounds__(256) void gemm_bf16_k(
    const unsigned short* __restrict__ A, int lda, long long aOuter, long long aInner,
    const unsigned short* __restrict__ Bm, int ldb, long long bOuter, long long bInner,
    float* __restrict__ Cf, unsigned short* __restrict__ Cb, int ldc,
    long long cOuter, long long cInner, int zdiv,
    const float* __restrict__ bias,
    const float* __restrict__ rowScale, int rsStride,
    float alpha, int M, int N, int K) {

    constexpr int NF = BN / 32;               // N-frags per wave
    constexpr int HN = BN / 2;                // columns per wave
    constexpr int CH = BN / 8;                // 8-elem chunks per k-row (non-trans B)
    constexpr int NT_ITERS = (32 * CH) / 256 > 0 ? (32 * CH) / 256 : 1;
    __shared__ unsigned short As[2][128 * 40];
    __shared__ unsigned short Bs[2][BN * 40];

    const int tid  = threadIdx.x;
    const int lane = tid & 31;
    const int wave = tid >> 5;
    const int wm = wave & 3;                  // 0..3 -> 32 rows each
    const int wn = wave >> 2;                 // 0..1 -> HN cols each
    const int m0 = blockIdx.y * 128;
    const int n0 = blockIdx.x * BN;
    const int z  = blockIdx.z;

    const long long aOff = (long long)(z / zdiv) * aOuter + (long long)(z % zdiv) * aInner;
    const long long bOff = (long long)(z / zdiv) * bOuter + (long long)(z % zdiv) * bInner;
    const long long cOff = (long long)(z / zdiv) * cOuter + (long long)(z % zdiv) * cInner;

    const int row16 = lane & 15;
    const int kbase = (lane >> 4) << 3;       // 0 or 8

    f32x8 acc[2][NF];
    f32x8 zero = {0.f, 0.f, 0.f, 0.f, 0.f, 0.f, 0.f, 0.f};
#pragma unroll
    for (int mf = 0; mf < 2; ++mf)
#pragma unroll
        for (int nf = 0; nf < NF; ++nf) acc[mf][nf] = zero;

    // --- staging helpers ---
    auto stageA = [&](int k0, int buf) {
        int row = tid >> 1, half = tid & 1;   // 256 threads cover 128 rows x 2 halves
        const unsigned short* gp = A + aOff + (size_t)(m0 + row) * lda + k0 + half * 16;
        copy32_g2l(gp, &As[buf][row * 40 + half * 16]);
    };
    auto stageB_t = [&](int k0, int buf) {    // transposed B: contiguous rows like A
        for (int idx = tid; idx < BN * 2; idx += 256) {
            int row = idx >> 1, half = idx & 1;
            const unsigned short* gp = Bm + bOff + (size_t)(n0 + row) * ldb + k0 + half * 16;
            copy32_g2l(gp, &Bs[buf][row * 40 + half * 16]);
        }
    };
    auto stageB_nt_load = [&](int k0, U128* regs) {   // non-trans B: gather to regs
#pragma unroll
        for (int it = 0; it < NT_ITERS; ++it) {
            int idx = tid + it * 256;
            int kk = idx / CH, nch = (idx % CH) * 8;
            regs[it] = *(const U128*)(Bm + bOff + (size_t)(k0 + kk) * ldb + n0 + nch);
        }
    };
    auto stageB_nt_store = [&](const U128* regs, int buf) {  // scatter regs -> LDS (transpose)
#pragma unroll
        for (int it = 0; it < NT_ITERS; ++it) {
            int idx = tid + it * 256;
            int kk = idx / CH, nch = (idx % CH) * 8;
            const unsigned short* ds = (const unsigned short*)&regs[it];
#pragma unroll
            for (int i = 0; i < 8; ++i) Bs[buf][(nch + i) * 40 + kk] = ds[i];
        }
    };
    auto compute = [&](int buf) {
        bf16x16 af[2];
#pragma unroll
        for (int mf = 0; mf < 2; ++mf)
            af[mf] = frag_ld(&As[buf][(wm * 32 + mf * 16 + row16) * 40 + kbase]);
#pragma unroll
        for (int nf = 0; nf < NF; ++nf) {
            bf16x16 bfr = frag_ld(&Bs[buf][(wn * HN + nf * 16 + row16) * 40 + kbase]);
#pragma unroll
            for (int mf = 0; mf < 2; ++mf)
                acc[mf][nf] = __builtin_amdgcn_wmma_f32_16x16x32_bf16(
                    false, af[mf], false, bfr, (short)0, acc[mf][nf], false, false);
        }
    };

    // --- prologue: stage tile 0 ---
    const int nk = K >> 5;
    stageA(0, 0);
    if constexpr (TRANSB) {
        stageB_t(0, 0);
    } else {
        U128 r0[NT_ITERS];
        stageB_nt_load(0, r0);
        stageB_nt_store(r0, 0);
    }
    wait_async_copies();
    __syncthreads();

    // --- pipelined main loop ---
    for (int t = 0; t < nk; ++t) {
        const int cur = t & 1, nxt = cur ^ 1;
        const bool have_next = (t + 1) < nk;
        U128 regs[NT_ITERS];
        if (have_next) {
            const int kn = (t + 1) << 5;
            stageA(kn, nxt);                         // async (or sync fallback)
            if constexpr (TRANSB) stageB_t(kn, nxt); // async (or sync fallback)
            else stageB_nt_load(kn, regs);           // global loads issue now...
            if (t + 2 < nk)                          // far prefetch of A tile t+2
                __builtin_prefetch(A + aOff + (size_t)(m0 + (tid >> 1)) * lda +
                                   ((t + 2) << 5) + (tid & 1) * 16, 0, 1);
        }
        compute(cur);                                // ...WMMAs overlap the copies
        if constexpr (!TRANSB) {
            if (have_next) stageB_nt_store(regs, nxt);
        }
        if (have_next) wait_async_copies();
        __syncthreads();
    }

    // ---- epilogue
#pragma unroll
    for (int mf = 0; mf < 2; ++mf) {
#pragma unroll
        for (int nf = 0; nf < NF; ++nf) {
#pragma unroll
            for (int r = 0; r < 8; ++r) {
                int gm = m0 + wm * 32 + mf * 16 + ((lane >> 4) << 3) + r;
                int gn = n0 + wn * HN + nf * 16 + (lane & 15);
                float v = alpha * acc[mf][nf][r];
                if (bias) v += bias[gn];
                if (GELU) v = 0.5f * v * (1.0f + erff(v * 0.7071067811865476f));
                if (rowScale) v *= rowScale[(size_t)gm * rsStride];
                size_t ci = (size_t)cOff + (size_t)gm * ldc + gn;
                if (STOREBF) {
                    Cb[ci] = f2bf(v);
                } else {
                    if (ACCUM) Cf[ci] += v; else Cf[ci] = v;
                }
            }
        }
    }
}

// ---------------- in-place causal softmax over bf16 score rows ----------------
__global__ __launch_bounds__(256) void softmax_causal_k(unsigned short* __restrict__ scores) {
    const int q   = blockIdx.x;
    const int z   = blockIdx.y;
    const int tid = threadIdx.x;
    unsigned short* row = scores + ((size_t)z * cS + q) * (size_t)cS;
    __shared__ float red[256];
    const int valid = q + 1;
    float mx = -3.0e38f;
    for (int c = tid; c < valid; c += 256) mx = fmaxf(mx, bf2f(row[c]));
    red[tid] = mx; __syncthreads();
    for (int s = 128; s > 0; s >>= 1) { if (tid < s) red[tid] = fmaxf(red[tid], red[tid + s]); __syncthreads(); }
    mx = red[0]; __syncthreads();
    float sum = 0.f;
    for (int c = tid; c < valid; c += 256) sum += __expf(bf2f(row[c]) - mx);
    red[tid] = sum; __syncthreads();
    for (int s = 128; s > 0; s >>= 1) { if (tid < s) red[tid] += red[tid + s]; __syncthreads(); }
    const float inv = 1.0f / red[0];
    for (int c = tid; c < cS; c += 256) {
        float p = (c < valid) ? __expf(bf2f(row[c]) - mx) * inv : 0.0f;
        row[c] = f2bf(p);
    }
}

// ---------------- residual + LayerNorm (writes f32 and bf16 copies) ----------------
__global__ __launch_bounds__(256) void residual_ln_k(const float* __restrict__ X,
                                                     const float* __restrict__ Aa,
                                                     const float* __restrict__ g,
                                                     const float* __restrict__ b,
                                                     float* __restrict__ Yf,
                                                     unsigned short* __restrict__ Ybf) {
    const int rowi = blockIdx.x;
    const int tid  = threadIdx.x;
    __shared__ float buf[cD];
    __shared__ float red[256];
    const float* x = X  + (size_t)rowi * cD;
    const float* a = Aa + (size_t)rowi * cD;
    float s = 0.f;
    for (int d = tid; d < cD; d += 256) { float v = x[d] + a[d]; buf[d] = v; s += v; }
    red[tid] = s; __syncthreads();
    for (int st = 128; st > 0; st >>= 1) { if (tid < st) red[tid] += red[tid + st]; __syncthreads(); }
    const float mean = red[0] / cD; __syncthreads();
    float vs = 0.f;
    for (int d = tid; d < cD; d += 256) { float t = buf[d] - mean; vs += t * t; }
    red[tid] = vs; __syncthreads();
    for (int st = 128; st > 0; st >>= 1) { if (tid < st) red[tid] += red[tid + st]; __syncthreads(); }
    const float rstd = rsqrtf(red[0] / cD + 1e-5f);
    for (int d = tid; d < cD; d += 256) {
        float y = (buf[d] - mean) * rstd * g[d] + b[d];
        Yf[(size_t)rowi * cD + d]  = y;
        Ybf[(size_t)rowi * cD + d] = f2bf(y);
    }
}

// ---------------- router: softmax over 8 experts, top-2, normalized weights ----------------
__global__ __launch_bounds__(256) void router_k(const float* __restrict__ X,
                                                const float* __restrict__ rw,
                                                const float* __restrict__ rb,
                                                float* __restrict__ mask) {
    const int t   = blockIdx.x;
    const int tid = threadIdx.x;
    const float* x = X + (size_t)t * cD;
    float loc[cE];
#pragma unroll
    for (int e = 0; e < cE; ++e) loc[e] = 0.f;
    for (int d = tid; d < cD; d += 256) {
        float xv = x[d];
#pragma unroll
        for (int e = 0; e < cE; ++e) loc[e] += xv * rw[e * cD + d];
    }
    __shared__ float red[256];
    __shared__ float logits[cE];
    for (int e = 0; e < cE; ++e) {
        red[tid] = loc[e]; __syncthreads();
        for (int st = 128; st > 0; st >>= 1) { if (tid < st) red[tid] += red[tid + st]; __syncthreads(); }
        if (tid == 0) logits[e] = red[0] + rb[e];
        __syncthreads();
    }
    if (tid == 0) {
        float mx = logits[0];
        for (int e = 1; e < cE; ++e) mx = fmaxf(mx, logits[e]);
        float p[cE], s = 0.f;
        for (int e = 0; e < cE; ++e) { p[e] = __expf(logits[e] - mx); s += p[e]; }
        for (int e = 0; e < cE; ++e) p[e] /= s;
        int i1 = 0;
        for (int e = 1; e < cE; ++e) if (p[e] > p[i1]) i1 = e;          // lowest index on tie
        int i2 = -1;
        for (int e = 0; e < cE; ++e) { if (e == i1) continue; if (i2 < 0 || p[e] > p[i2]) i2 = e; }
        float wsum = p[i1] + p[i2];
        float wa = p[i1] / wsum, wb = p[i2] / wsum;
        for (int e = 0; e < cE; ++e) mask[(size_t)t * cE + e] = 0.0f;
        mask[(size_t)t * cE + i1] = wa;
        mask[(size_t)t * cE + i2] = wb;
    }
}

// ---------------- aux loss from routing mask (deterministic fixed-order reduction) ----------------
__global__ __launch_bounds__(256) void aux_from_mask_k(const float* __restrict__ mask,
                                                       float* __restrict__ auxOut) {
    const int tid = threadIdx.x;
    float cnt[cE], ps[cE];
#pragma unroll
    for (int e = 0; e < cE; ++e) { cnt[e] = 0.f; ps[e] = 0.f; }
    for (int t = tid; t < TOKENS; t += 256) {
#pragma unroll
        for (int e = 0; e < cE; ++e) {
            float mv = mask[(size_t)t * cE + e];
            ps[e] += mv;
            if (mv > 0.f) cnt[e] += 1.f;
        }
    }
    __shared__ float red[256];
    __shared__ float fc[cE], fp[cE];
    for (int e = 0; e < cE; ++e) {
        red[tid] = cnt[e]; __syncthreads();
        for (int st = 128; st > 0; st >>= 1) { if (tid < st) red[tid] += red[tid + st]; __syncthreads(); }
        if (tid == 0) fc[e] = red[0];
        __syncthreads();
        red[tid] = ps[e]; __syncthreads();
        for (int st = 128; st > 0; st >>= 1) { if (tid < st) red[tid] += red[tid + st]; __syncthreads(); }
        if (tid == 0) fp[e] = red[0];
        __syncthreads();
    }
    if (tid == 0) {
        float usum = 0.f;
        for (int e = 0; e < cE; ++e) usum += fc[e];
        float s = 0.f;
        for (int e = 0; e < cE; ++e) s += (fc[e] / usum) * (fp[e] / (float)TOKENS);
        *auxOut = (float)cE * s * 0.01f;
    }
}

__global__ void finalize_aux_k(const float* __restrict__ auxbuf, float* __restrict__ out) {
    if (threadIdx.x == 0 && blockIdx.x == 0) out[0] = auxbuf[0] + auxbuf[1];
}

// ---------------- host orchestration ----------------
extern "C" void kernel_launch(void* const* d_in, const int* in_sizes, int n_in,
                              void* d_out, int out_size, void* d_ws, size_t ws_size,
                              hipStream_t stream) {
    (void)in_sizes; (void)n_in; (void)out_size; (void)ws_size;
    const float* tgt      = (const float*)d_in[0];
    const float* qkv_w    = (const float*)d_in[1];
    const float* qkv_b    = (const float*)d_in[2];
    const float* out_w    = (const float*)d_in[3];
    const float* out_b    = (const float*)d_in[4];
    const float* ln1_g    = (const float*)d_in[5];
    const float* ln1_b    = (const float*)d_in[6];
    const float* ln2_g    = (const float*)d_in[7];
    const float* ln2_b    = (const float*)d_in[8];
    const float* router_w = (const float*)d_in[9];
    const float* router_b = (const float*)d_in[10];
    const float* w1       = (const float*)d_in[11];
    const float* b1       = (const float*)d_in[12];
    const float* w2       = (const float*)d_in[13];
    const float* b2       = (const float*)d_in[14];
    float* outp = (float*)d_out;

    // bump allocator over workspace
    char* base = (char*)d_ws;
    size_t off = 0;
    auto alloc = [&](size_t bytes) -> void* {
        void* p = base + off;
        off = (off + bytes + 255) & ~(size_t)255;
        return p;
    };
    unsigned short* qkvw_bf  = (unsigned short*)alloc(sizeof(short) * (size_t)cL * 3 * cD * cD);
    unsigned short* outw_bf  = (unsigned short*)alloc(sizeof(short) * (size_t)cL * cD * cD);
    unsigned short* w1_bf    = (unsigned short*)alloc(sizeof(short) * (size_t)cL * cE * cD * cFF);
    unsigned short* w2_bf    = (unsigned short*)alloc(sizeof(short) * (size_t)cL * cE * cFF * cD);
    float*          x_cur    = (float*)alloc(sizeof(float) * (size_t)BSD);
    unsigned short* x_bf     = (unsigned short*)alloc(sizeof(short) * (size_t)BSD);
    unsigned short* qkv_bf   = (unsigned short*)alloc(sizeof(short) * (size_t)TOKENS * 3 * cD);
    unsigned short* sc_bf    = (unsigned short*)alloc(sizeof(short) * (size_t)cB * cH * cS * cS);
    unsigned short* ctx_bf   = (unsigned short*)alloc(sizeof(short) * (size_t)BSD);
    float*          attnout  = (float*)alloc(sizeof(float) * (size_t)BSD);
    unsigned short* h_bf     = (unsigned short*)alloc(sizeof(short) * (size_t)TOKENS * cFF);
    float*          ymoe     = (float*)alloc(sizeof(float) * (size_t)BSD);
    float*          maskbuf  = (float*)alloc(sizeof(float) * (size_t)TOKENS * cE);
    float*          auxbuf   = (float*)alloc(sizeof(float) * cL);

    // ---- per-launch f32 -> bf16 weight conversions
    cvt_f32_to_bf16_k<<<4096, 256, 0, stream>>>(qkv_w, qkvw_bf, (size_t)cL * 3 * cD * cD);
    cvt_f32_to_bf16_k<<<4096, 256, 0, stream>>>(out_w, outw_bf, (size_t)cL * cD * cD);
    cvt_f32_to_bf16_k<<<4096, 256, 0, stream>>>(w1, w1_bf, (size_t)cL * cE * cD * cFF);
    cvt_f32_to_bf16_k<<<4096, 256, 0, stream>>>(w2, w2_bf, (size_t)cL * cE * cFF * cD);
    cvt_f32_to_bf16_k<<<2048, 256, 0, stream>>>(tgt, x_bf, (size_t)BSD);

    for (int l = 0; l < cL; ++l) {
        const unsigned short* qkvw_l = qkvw_bf + (size_t)l * 3 * cD * cD;
        const unsigned short* outw_l = outw_bf + (size_t)l * cD * cD;
        const float* qkvb_l = qkv_b + (size_t)l * 3 * cD;
        const float* outb_l = out_b + (size_t)l * cD;
        const unsigned short* w1_l = w1_bf + (size_t)l * cE * cD * cFF;
        const unsigned short* w2_l = w2_bf + (size_t)l * cE * cFF * cD;
        const float* b1_l = b1 + (size_t)l * cE * cFF;
        const float* b2_l = b2 + (size_t)l * cE * cD;

        // 1) QKV projection: [2048,1024] @ [1024,3072] (W^T) -> bf16 qkv
        gemm_bf16_k<128, true, false, true, false><<<dim3(3 * cD / 128, TOKENS / 128, 1), 256, 0, stream>>>(
            x_bf, cD, 0, 0, qkvw_l, cD, 0, 0,
            nullptr, qkv_bf, 3 * cD, 0, 0, 1,
            qkvb_l, nullptr, 0, 1.0f, TOKENS, 3 * cD, cD);

        // 2) scores = Q @ K^T / sqrt(dh), batched over z = b*H + h (zdiv = H)
        gemm_bf16_k<128, true, false, true, false><<<dim3(cS / 128, cS / 128, cB * cH), 256, 0, stream>>>(
            qkv_bf, 3 * cD, (long long)cS * 3 * cD, DH,
            qkv_bf + cD, 3 * cD, (long long)cS * 3 * cD, DH,
            nullptr, sc_bf, cS, (long long)cH * cS * cS, (long long)cS * cS, cH,
            nullptr, nullptr, 0, 0.125f, cS, cS, DH);

        // 3) causal softmax, in place on bf16 scores
        softmax_causal_k<<<dim3(cS, cB * cH), 256, 0, stream>>>(sc_bf);

        // 4) ctx = P @ V (N = dh = 64 -> BN=64 tile)
        gemm_bf16_k<64, false, false, true, false><<<dim3(1, cS / 128, cB * cH), 256, 0, stream>>>(
            sc_bf, cS, (long long)cH * cS * cS, (long long)cS * cS,
            qkv_bf + 2 * cD, 3 * cD, (long long)cS * 3 * cD, DH,
            nullptr, ctx_bf, cD, (long long)cS * cD, DH, cH,
            nullptr, nullptr, 0, 1.0f, cS, DH, cS);

        // 5) attention output projection -> f32
        gemm_bf16_k<128, true, false, false, false><<<dim3(cD / 128, TOKENS / 128, 1), 256, 0, stream>>>(
            ctx_bf, cD, 0, 0, outw_l, cD, 0, 0,
            attnout, nullptr, cD, 0, 0, 1,
            outb_l, nullptr, 0, 1.0f, TOKENS, cD, cD);

        // 6) x = LN1(x + attn)
        residual_ln_k<<<TOKENS, 256, 0, stream>>>(
            (l == 0) ? tgt : x_cur, attnout,
            ln1_g + (size_t)l * cD, ln1_b + (size_t)l * cD, x_cur, x_bf);

        // 7) router -> normalized top-2 mask; aux loss for this layer
        router_k<<<TOKENS, 256, 0, stream>>>(x_cur, router_w + (size_t)l * cE * cD,
                                             router_b + (size_t)l * cE, maskbuf);
        aux_from_mask_k<<<1, 256, 0, stream>>>(maskbuf, auxbuf + l);

        // 8) dense-dispatch MoE: loop experts; e==0 writes, e>0 accumulates
        for (int e = 0; e < cE; ++e) {
            // h = gelu(x @ W1[e] + b1[e]) -> bf16
            gemm_bf16_k<128, false, true, true, false><<<dim3(cFF / 128, TOKENS / 128, 1), 256, 0, stream>>>(
                x_bf, cD, 0, 0, w1_l + (size_t)e * cD * cFF, cFF, 0, 0,
                nullptr, h_bf, cFF, 0, 0, 1,
                b1_l + (size_t)e * cFF, nullptr, 0, 1.0f, TOKENS, cFF, cD);
            // ymoe (+)= mask[:,e] * (h @ W2[e] + b2[e])
            if (e == 0) {
                gemm_bf16_k<128, false, false, false, false><<<dim3(cD / 128, TOKENS / 128, 1), 256, 0, stream>>>(
                    h_bf, cFF, 0, 0, w2_l + (size_t)e * cFF * cD, cD, 0, 0,
                    ymoe, nullptr, cD, 0, 0, 1,
                    b2_l + (size_t)e * cD, maskbuf + e, cE, 1.0f, TOKENS, cD, cFF);
            } else {
                gemm_bf16_k<128, false, false, false, true><<<dim3(cD / 128, TOKENS / 128, 1), 256, 0, stream>>>(
                    h_bf, cFF, 0, 0, w2_l + (size_t)e * cFF * cD, cD, 0, 0,
                    ymoe, nullptr, cD, 0, 0, 1,
                    b2_l + (size_t)e * cD, maskbuf + e, cE, 1.0f, TOKENS, cD, cFF);
            }
        }

        // 9) x = LN2(x + moe); last layer writes f32 result straight into d_out
        residual_ln_k<<<TOKENS, 256, 0, stream>>>(
            x_cur, ymoe, ln2_g + (size_t)l * cD, ln2_b + (size_t)l * cD,
            (l == cL - 1) ? outp : x_cur, x_bf);
    }

    // aux_total -> d_out[B*S*D]
    finalize_aux_k<<<1, 32, 0, stream>>>(auxbuf, outp + BSD);
}
